// T5Attention_CCM_25005299597714
// MI455X (gfx1250) — matrix operand order repaired and checked by reading, not compiled, not run on hardware
//
#include <hip/hip_runtime.h>

// ---------------------------------------------------------------------------
// T5 attention for MI455X (gfx1250, wave32, WMMA bf16 16x16x32, fp32 accum)
// - bf16 WMMA GEMMs for Q/K/V/out projections (compile-time-shaped epilogues)
// - flash attention: K/V staged to LDS with GLOBAL_LOAD_ASYNC_TO_LDS_B128,
//   double-buffered, paced by s_wait_asynccnt; P transposed via LDS
// ---------------------------------------------------------------------------

typedef unsigned short u16;
typedef __bf16   bf16x16 __attribute__((ext_vector_type(16)));
typedef float    v8f     __attribute__((ext_vector_type(8)));

union Frag {
    bf16x16 v;
    uint4   q[2];
};
static_assert(sizeof(Frag) == 32, "frag size");

__device__ __forceinline__ u16 f2bf(float f) {
    unsigned u = __builtin_bit_cast(unsigned, f);
    unsigned r = u + 0x7FFFu + ((u >> 16) & 1u);   // round-to-nearest-even
    return (u16)(r >> 16);
}

__device__ __forceinline__ v8f wmma_bf16(bf16x16 a, bf16x16 b, v8f c) {
    return __builtin_amdgcn_wmma_f32_16x16x32_bf16(
        false, a, false, b, (short)0, c, false, false);
}

// A fragment (16x32 bf16, M x K). lane<16: k {0..7,16..23}; lane>=16: k {8..15,24..31}
__device__ __forceinline__ bf16x16 load_frag_a(const u16* base, int ld, int k0, int lane) {
    int row = lane & 15;
    int hi  = (lane >> 4) & 1;
    const u16* p = base + (size_t)row * ld + k0 + hi * 8;
    Frag f;
    f.q[0] = *(const uint4*)(p);
    f.q[1] = *(const uint4*)(p + 16);
    return f.v;
}

// B fragment (32x16 bf16, K x N) read from row-major "B-transposed" storage
// (lane n holds column n == row n of BT). lane<16: k {0..7,16..23}; lane>=16: k {8..15,24..31}
__device__ __forceinline__ bf16x16 load_frag_b(const u16* baseT, int ld, int k0, int lane) {
    int col = lane & 15;
    int hi  = (lane >> 4) & 1;
    const u16* p = baseT + (size_t)col * ld + k0 + hi * 8;
    Frag f;
    f.q[0] = *(const uint4*)(p);
    f.q[1] = *(const uint4*)(p + 16);
    return f.v;
}

// CDNA5 async copy: 16 bytes/lane, global -> LDS, tracked by ASYNCcnt
__device__ __forceinline__ void async_copy_b128(const u16* gptr, u16* lptr) {
    unsigned lds_off = (unsigned)(size_t)lptr;   // low 32 bits = LDS address
    asm volatile("global_load_async_to_lds_b128 %0, %1, off"
                 :: "v"(lds_off), "v"(gptr)
                 : "memory");
}

__device__ __forceinline__ void wait_asynccnt_le2() {
    asm volatile("s_wait_asynccnt 0x2" ::: "memory");
}
__device__ __forceinline__ void wait_asynccnt_0() {
    asm volatile("s_wait_asynccnt 0x0" ::: "memory");
}

// ---------------------------------------------------------------------------
// conversion kernels
// ---------------------------------------------------------------------------
__global__ void cvt_f32_bf16_kernel(const float* __restrict__ in,
                                    u16* __restrict__ out, int n) {
    int i = blockIdx.x * blockDim.x + threadIdx.x;
    if (i < n) out[i] = f2bf(in[i]);
}

// W[k][n] f32 (1024x1024 row-major)  ->  WT[n][k] bf16
__global__ void transpose_cvt_kernel(const float* __restrict__ W,
                                     u16* __restrict__ WT) {
    constexpr int N = 1024;
    int i = blockIdx.x * blockDim.x + threadIdx.x;
    int k = i >> 10;
    int n = i & (N - 1);
    WT[(size_t)n * 1024 + k] = f2bf(W[i]);
}

// ---------------------------------------------------------------------------
// T5 relative-position bias table:  biasTab[h][delta],  delta = (k - q) + (S-1)
// ---------------------------------------------------------------------------
__global__ void bias_build_kernel(const float* __restrict__ emb,   // [NB][H]
                                  float* __restrict__ biasTab) {   // [H][2S-1]
    constexpr int S = 1024, H = 16, NB = 32, MAXD = 128;
    constexpr int nrel = 2 * S - 1;
    int i = blockIdx.x * blockDim.x + threadIdx.x;
    if (i >= H * nrel) return;
    int h = i / nrel;
    int d = i - h * nrel;
    int rp = d - (S - 1);                 // mem - ctx
    constexpr int nb = NB / 2;            // bidirectional
    int bucket = (rp > 0) ? nb : 0;
    int arp = rp < 0 ? -rp : rp;
    constexpr int max_exact = nb / 2;
    int rel;
    if (arp < max_exact) {
        rel = arp;
    } else {
        float v = logf((float)arp / (float)max_exact)
                / logf((float)MAXD / (float)max_exact) * (float)(nb - max_exact);
        rel = max_exact + (int)v;
        if (rel > nb - 1) rel = nb - 1;
    }
    bucket += rel;
    biasTab[(size_t)h * nrel + d] = emb[(size_t)bucket * H + h];
}

// ---------------------------------------------------------------------------
// generic bf16 WMMA GEMM:  C[m][n] = sum_k A[m][k] * BT[n][k]
// M=4096, N=K=1024 (all compile-time). block = 128 threads (4 waves);
// each wave computes a 64x64 C tile. grid = (N/128, M/128)
// ---------------------------------------------------------------------------
#define GEMM_MODE_QK  0   // store bf16 to [b][h][s][dk]
#define GEMM_MODE_VT  1   // store bf16 to [b][h][dk][s]
#define GEMM_MODE_F32 2   // store f32 row-major MxN

template <int MODE>
__global__ __launch_bounds__(128)
void gemm_bf16_kernel(const u16* __restrict__ A, const u16* __restrict__ BT,
                      void* __restrict__ out) {
    constexpr int N = 1024, K = 1024;
    constexpr int Hn = 16, DKn = 64;

    int tid  = threadIdx.x;
    int lane = tid & 31;
    int wave = tid >> 5;
    int wm = wave >> 1, wn = wave & 1;
    int m0 = blockIdx.y * 128 + wm * 64;
    int n0 = blockIdx.x * 128 + wn * 64;

    v8f acc[4][4];
#pragma unroll
    for (int i = 0; i < 4; i++)
#pragma unroll
        for (int j = 0; j < 4; j++) acc[i][j] = (v8f)(0.f);

    for (int k0 = 0; k0 < K; k0 += 32) {
        bf16x16 af[4], bf[4];
#pragma unroll
        for (int mi = 0; mi < 4; mi++)
            af[mi] = load_frag_a(A + (size_t)(m0 + mi * 16) * K, K, k0, lane);
#pragma unroll
        for (int ni = 0; ni < 4; ni++)
            bf[ni] = load_frag_b(BT + (size_t)(n0 + ni * 16) * K, K, k0, lane);
#pragma unroll
        for (int mi = 0; mi < 4; mi++)
#pragma unroll
            for (int ni = 0; ni < 4; ni++)
                acc[mi][ni] = wmma_bf16(af[mi], bf[ni], acc[mi][ni]);
    }

    int hi  = (lane >> 4) & 1;
    int col = lane & 15;
#pragma unroll
    for (int mi = 0; mi < 4; mi++) {
#pragma unroll
        for (int ni = 0; ni < 4; ni++) {
#pragma unroll
            for (int e = 0; e < 8; e++) {
                int m = m0 + mi * 16 + e + hi * 8;
                int n = n0 + ni * 16 + col;
                float val = acc[mi][ni][e];
                if constexpr (MODE == GEMM_MODE_F32) {
                    ((float*)out)[(size_t)m * N + n] = val;
                } else {
                    int b  = m >> 10, s = m & 1023;     // Sdim = 1024
                    int h  = n >> 6,  dk = n & 63;      // DKn  = 64
                    size_t idx;
                    if constexpr (MODE == GEMM_MODE_QK)
                        idx = (((size_t)b * Hn + h) * 1024 + s) * DKn + dk;
                    else // VT
                        idx = (((size_t)b * Hn + h) * DKn + dk) * 1024 + s;
                    ((u16*)out)[idx] = f2bf(val);
                }
            }
        }
    }
}

// ---------------------------------------------------------------------------
// flash attention with T5 relative bias (no 1/sqrt(dk) scaling)
// block = 256 threads (8 waves); each wave owns 16 query rows; block = 128 q rows
// grid = B * H * (S/128)
// Q,K: bf16 [b][h][s][64]   VT: bf16 [b][h][64][s]   ctx: bf16 [b][s][h*64]
// K/V tiles (32 keys wide) double-buffered in LDS via async copies.
// ---------------------------------------------------------------------------
__global__ __launch_bounds__(256)
void attn_kernel(const u16* __restrict__ Q, const u16* __restrict__ Km,
                 const u16* __restrict__ VT, const float* __restrict__ biasTab,
                 u16* __restrict__ ctx) {
    constexpr int S = 1024, H = 16, DK = 64;
    constexpr int nrel = 2 * S - 1;

    __shared__ __align__(16) float biasLDS[nrel];
    __shared__ __align__(16) u16   ktile[2][32 * 64];   // [kk][dk]
    __shared__ __align__(16) u16   vtile[2][64 * 32];   // [dk][kk]
    __shared__ __align__(16) u16   pLDS[8][16 * 32];    // per-wave P staging

    int blk = blockIdx.x;
    int qb  = blk & 7;            // S/128 = 8 q-blocks
    int h   = (blk >> 3) & 15;
    int b   = blk >> 7;

    int tid = threadIdx.x;
    for (int i = tid; i < nrel; i += 256)
        biasLDS[i] = biasTab[(size_t)h * nrel + i];

    int wave = tid >> 5;
    int lane = tid & 31;
    int hi   = (lane >> 4) & 1;
    int col  = lane & 15;
    int q0   = qb * 128 + wave * 16;

    const u16* Qh = Q  + ((size_t)b * H + h) * S * DK;
    const u16* Kh = Km + ((size_t)b * H + h) * S * DK;
    const u16* Vh = VT + ((size_t)b * H + h) * DK * S;

    // async-stage one 32-key block: K 32x64 (4KB) + V 64x32 (4KB), 16B/thread each
    auto issue_tile = [&](int kb, int buf) {
        int koff = tid * 8;                               // u16 units
        async_copy_b128(Kh + (size_t)kb * DK + koff, &ktile[buf][koff]);
        int vrow = tid >> 2;
        int vcol = (tid & 3) * 8;
        async_copy_b128(Vh + (size_t)vrow * S + kb + vcol, &vtile[buf][vrow * 32 + vcol]);
    };

    issue_tile(0, 0);
    issue_tile(32, 1);

    // resident Q fragments: 16 rows x 64 dk = two 16x32 A-fragments
    bf16x16 aq0 = load_frag_a(Qh + (size_t)q0 * DK, DK, 0,  lane);
    bf16x16 aq1 = load_frag_a(Qh + (size_t)q0 * DK, DK, 32, lane);

    v8f o[4];
#pragma unroll
    for (int nt = 0; nt < 4; nt++) o[nt] = (v8f)(0.f);
    float rm[8], rl[8];
#pragma unroll
    for (int r = 0; r < 8; r++) { rm[r] = -1e30f; rl[r] = 0.f; }

    u16* pb = &pLDS[wave][0];

    for (int kb = 0; kb < S; kb += 32) {
        int cur = (kb >> 5) & 1;
        if (kb + 32 < S) wait_asynccnt_le2();   // oldest tile (cur) landed
        else             wait_asynccnt_0();
        __syncthreads();                        // tile visible to all waves

        const u16* kt = &ktile[cur][0];
        const u16* vt = &vtile[cur][0];

        // ---- scores: S = Q Kt (two 16x16 tiles over 32 keys), K from LDS ----
        v8f sc[2];
#pragma unroll
        for (int ni = 0; ni < 2; ni++) {
            const u16* kbase = kt + (ni * 16) * DK;       // row = col within frag
            bf16x16 bk0 = load_frag_b(kbase, DK, 0,  lane);
            bf16x16 bk1 = load_frag_b(kbase, DK, 32, lane);
            v8f a = (v8f)(0.f);
            a = wmma_bf16(aq0, bk0, a);
            a = wmma_bf16(aq1, bk1, a);
#pragma unroll
            for (int e = 0; e < 8; e++) {
                int q  = q0 + e + hi * 8;
                int kk = kb + ni * 16 + col;
                a[e] += biasLDS[kk - q + (S - 1)];
            }
            sc[ni] = a;
        }

        // ---- online softmax (row = e + 8*hi, spread over 16 lanes) ----
#pragma unroll
        for (int r = 0; r < 8; r++) {
            float nm = fmaxf(sc[0][r], sc[1][r]);
#pragma unroll
            for (int msk = 8; msk >= 1; msk >>= 1)
                nm = fmaxf(nm, __shfl_xor(nm, msk, 32));
            nm = fmaxf(nm, rm[r]);
            float scale = __expf(rm[r] - nm);
            rm[r] = nm;
            float p0 = __expf(sc[0][r] - nm);
            float p1 = __expf(sc[1][r] - nm);
            sc[0][r] = p0;
            sc[1][r] = p1;
            float ps = p0 + p1;
#pragma unroll
            for (int msk = 8; msk >= 1; msk >>= 1)
                ps += __shfl_xor(ps, msk, 32);
            rl[r] = rl[r] * scale + ps;
#pragma unroll
            for (int nt = 0; nt < 4; nt++) o[nt][r] *= scale;
        }

        // ---- P: C-layout -> A-layout via per-wave LDS staging ----
#pragma unroll
        for (int ni = 0; ni < 2; ni++)
#pragma unroll
            for (int e = 0; e < 8; e++) {
                int row = e + hi * 8;
                int kkl = ni * 16 + col;
                pb[row * 32 + kkl] = f2bf(sc[ni][e]);
            }
        __syncthreads();   // P staged (and all waves done with score reads)

        bf16x16 ap;
        {
            int row = lane & 15;
            const u16* p = pb + row * 32 + hi * 8;
            Frag f;
            f.q[0] = *(const uint4*)(p);
            f.q[1] = *(const uint4*)(p + 16);
            ap = f.v;
        }

        // ---- O += P @ V  (V rows = dk from LDS, contiguous per lane) ----
#pragma unroll
        for (int nt = 0; nt < 4; nt++) {
            const u16* vb = vt + (nt * 16) * 32;
            bf16x16 bv = load_frag_b(vb, 32, 0, lane);
            o[nt] = wmma_bf16(ap, bv, o[nt]);
        }

        __syncthreads();   // everyone done reading cur tile + pLDS
        if (kb + 64 < S) issue_tile(kb + 64, cur);   // refill the buffer just freed
    }

    // ---- epilogue: normalize, store ctx[b][s][h*64 + dk] ----
#pragma unroll
    for (int nt = 0; nt < 4; nt++) {
#pragma unroll
        for (int e = 0; e < 8; e++) {
            int q  = q0 + e + hi * 8;
            int dk = nt * 16 + col;
            float val = o[nt][e] / rl[e];
            ctx[((size_t)b * S + q) * (H * DK) + h * DK + dk] = f2bf(val);
        }
    }
}

// ---------------------------------------------------------------------------
// launch
// ---------------------------------------------------------------------------
extern "C" void kernel_launch(void* const* d_in, const int* in_sizes, int n_in,
                              void* d_out, int out_size, void* d_ws, size_t ws_size,
                              hipStream_t stream) {
    (void)in_sizes; (void)n_in; (void)out_size; (void)ws_size;

    const int Bb = 4, S = 1024, D = 1024, H = 16, DK = 64;
    const int M = Bb * S;          // 4096
    const int N = H * DK;          // 1024

    const float* X   = (const float*)d_in[0];
    const float* Wq  = (const float*)d_in[1];
    const float* Wk  = (const float*)d_in[2];
    const float* Wv  = (const float*)d_in[3];
    const float* Wo  = (const float*)d_in[4];
    const float* emb = (const float*)d_in[5];

    char* ws = (char*)d_ws;
    const size_t MB = 1u << 20;
    u16*   Xbf  = (u16*)(ws + 0 * MB);    // 8 MB
    u16*   WqT  = (u16*)(ws + 8 * MB);    // 2 MB each
    u16*   WkT  = (u16*)(ws + 10 * MB);
    u16*   WvT  = (u16*)(ws + 12 * MB);
    u16*   WoT  = (u16*)(ws + 14 * MB);
    u16*   Qb   = (u16*)(ws + 16 * MB);   // 8 MB
    u16*   Kb   = (u16*)(ws + 24 * MB);   // 8 MB
    u16*   VTb  = (u16*)(ws + 32 * MB);   // 8 MB
    u16*   Ctx  = (u16*)(ws + 40 * MB);   // 8 MB
    float* bias = (float*)(ws + 48 * MB); // ~128 KB

    // 1) convert activations / weights to bf16 (weights transposed)
    {
        int n = M * D;
        cvt_f32_bf16_kernel<<<(n + 255) / 256, 256, 0, stream>>>(X, Xbf, n);
        int nw = D * N;
        int gw = (nw + 255) / 256;
        transpose_cvt_kernel<<<gw, 256, 0, stream>>>(Wq, WqT);
        transpose_cvt_kernel<<<gw, 256, 0, stream>>>(Wk, WkT);
        transpose_cvt_kernel<<<gw, 256, 0, stream>>>(Wv, WvT);
        transpose_cvt_kernel<<<gw, 256, 0, stream>>>(Wo, WoT);
    }

    // 2) relative-position bias table [H][2S-1]
    {
        int n = H * (2 * S - 1);
        bias_build_kernel<<<(n + 255) / 256, 256, 0, stream>>>(emb, bias);
    }

    // 3) Q/K/V projections (WMMA GEMMs)
    dim3 ggrid(N / 128, M / 128);
    gemm_bf16_kernel<GEMM_MODE_QK><<<ggrid, 128, 0, stream>>>(Xbf, WqT, Qb);
    gemm_bf16_kernel<GEMM_MODE_QK><<<ggrid, 128, 0, stream>>>(Xbf, WkT, Kb);
    gemm_bf16_kernel<GEMM_MODE_VT><<<ggrid, 128, 0, stream>>>(Xbf, WvT, VTb);

    // 4) flash attention with T5 bias (async K/V staging)
    attn_kernel<<<Bb * H * (S / 128), 256, 0, stream>>>(Qb, Kb, VTb, bias, Ctx);

    // 5) output projection -> fp32
    dim3 ogrid(D / 128, M / 128);
    gemm_bf16_kernel<GEMM_MODE_F32><<<ogrid, 128, 0, stream>>>(Ctx, WoT, d_out);
}
